// SelfAttention_45595372814366
// MI455X (gfx1250) — matrix operand order repaired and checked by reading, not compiled, not run on hardware
//
#include <hip/hip_runtime.h>

typedef __attribute__((ext_vector_type(16))) _Float16 v16h;
typedef __attribute__((ext_vector_type(8)))  _Float16 v8h;
typedef __attribute__((ext_vector_type(8)))  float    v8f;

union V16HU { v16h h; unsigned u[8]; };

#define WMMA_F16(A, B, C) \
  __builtin_amdgcn_wmma_f32_16x16x32_f16(false, (A), false, (B), (short)0, (C), false, false)

// Cross-half lane swap (lane <-> lane^16) without touching the LDS pipe.
static __device__ __forceinline__ unsigned xswap16_u(unsigned v) {
#if __has_builtin(__builtin_amdgcn_permlanex16)
  return (unsigned)__builtin_amdgcn_permlanex16((int)v, (int)v, 0x76543210, 0xfedcba98,
                                                false, false);
#else
  return (unsigned)__shfl_xor((int)v, 16, 32);
#endif
}
static __device__ __forceinline__ float xswap16_f(float v) {
  return __int_as_float((int)xswap16_u((unsigned)__float_as_int(v)));
}
static __device__ __forceinline__ unsigned pack2h(float a, float b) {
  union { _Float16 h[2]; unsigned u; } x;
  x.h[0] = (_Float16)a; x.h[1] = (_Float16)b;
  return x.u;
}

// A-fragment (16x32 f16, M x K): lane = row (lane&15); lanes 0-15 hold K 0-7 & 16-23,
// lanes 16-31 hold K 8-15 & 24-31. Source is row-major with row stride `ldr` halves.
static __device__ __forceinline__ v16h load_frag_a(const _Float16* base, int ldr, int lane) {
  const int row = lane & 15;
  const int kb  = (lane & 16) ? 8 : 0;
  const _Float16* p = base + (size_t)row * ldr + kb;
  v8h lo = *(const v8h*)(p);
  v8h hi = *(const v8h*)(p + 16);
  v16h a;
#pragma unroll
  for (int e = 0; e < 8; ++e) { a[e] = lo[e]; a[e + 8] = hi[e]; }
  return a;
}

// B-fragment (32x16 f16, K x N): lane = column; lanes 0-15 hold K 0-7 & 16-23,
// lanes 16-31 hold K 8-15 & 24-31. Caller passes per-lane pointer offset by K 0 / 8.
static __device__ __forceinline__ v16h load_frag_b(const _Float16* p) {
  v8h lo = *(const v8h*)(p);
  v8h hi = *(const v8h*)(p + 16);
  v16h b;
#pragma unroll
  for (int e = 0; e < 8; ++e) { b[e] = lo[e]; b[e + 8] = hi[e]; }
  return b;
}

// ---------------- Kernel 1: QKV projection (1x1 conv over channels) ----------------
// x:[B,64,N] f32 -> Qt (pre-scaled by log2(e)), Kt:[B,N,64] f16 ; Vf:[B,64,N] f16
__global__ __launch_bounds__(256) void qkv_proj(
    const float* __restrict__ x,
    const float* __restrict__ Wq, const float* __restrict__ bq,
    const float* __restrict__ Wk, const float* __restrict__ bk,
    const float* __restrict__ Wv, const float* __restrict__ bv,
    _Float16* __restrict__ Qt, _Float16* __restrict__ Kt, _Float16* __restrict__ Vf) {
  __shared__ float sW[3 * 64 * 64];
  __shared__ float sb[3 * 64];
  const int t = threadIdx.x;
  for (int idx = t; idx < 64 * 64; idx += 256) {
    sW[idx]        = Wq[idx];
    sW[4096 + idx] = Wk[idx];
    sW[8192 + idx] = Wv[idx];
  }
  if (t < 64) { sb[t] = bq[t]; sb[64 + t] = bk[t]; sb[128 + t] = bv[t]; }
  __syncthreads();

  const int b  = blockIdx.x >> 6;          // 64 tiles of 64 pixels per batch
  const int i0 = (blockIdx.x & 63) << 6;
  const int iL = t & 63;
  const int cq = t >> 6;                   // 0..3 -> 16 output channels each
  const int i  = i0 + iL;

  float aq[16], ak[16], av[16];
#pragma unroll
  for (int u = 0; u < 16; ++u) {
    const int c = cq * 16 + u;
    aq[u] = sb[c]; ak[u] = sb[64 + c]; av[u] = sb[128 + c];
  }
  const float* xp = x + ((size_t)b * 64) * 4096 + i;
#pragma unroll 4
  for (int cc = 0; cc < 64; ++cc) {
    const float xv = xp[(size_t)cc * 4096];
#pragma unroll
    for (int u = 0; u < 16; ++u) {
      const int c = cq * 16 + u;
      aq[u] = fmaf(sW[c * 64 + cc],        xv, aq[u]);
      ak[u] = fmaf(sW[4096 + c * 64 + cc], xv, ak[u]);
      av[u] = fmaf(sW[8192 + c * 64 + cc], xv, av[u]);
    }
  }
  const float LOG2E = 1.44269504088896341f;   // fold exp->exp2 conversion into Q
  const size_t rowQ = ((size_t)(b * 4096 + i)) * 64;
#pragma unroll
  for (int u = 0; u < 16; ++u) {
    const int c = cq * 16 + u;
    Qt[rowQ + c] = (_Float16)(aq[u] * LOG2E);
    Kt[rowQ + c] = (_Float16)ak[u];
    Vf[((size_t)(b * 64 + c)) * 4096 + i] = (_Float16)av[u];
  }
}

// ---------------- Kernel 2: fused flash attention (S^T orientation) ----------------
// S^T = K^T Q: each lane owns one query column -> per-lane scalar m/l, no LDS needed.
__global__ __launch_bounds__(256) void flash_attn(
    const _Float16* __restrict__ Qt, const _Float16* __restrict__ Kt,
    const _Float16* __restrict__ Vf, float* __restrict__ out) {
  const int lane = threadIdx.x & 31;
  const int wv   = threadIdx.x >> 5;
  const int tile = blockIdx.x * 8 + wv;                  // 0..1023
  const int b    = tile >> 8;
  const int i0   = (tile & 255) << 4;
  const int n    = lane & 15;                            // query column owned by lane
  const int hl   = lane >> 4;                            // lane-half
  const int koff = hl << 3;                              // B-frag K split (0 / 8)

  // All-ones A fragment: lsum = ones x P^T gives per-query sums replicated over rows.
  v16h ones;
#pragma unroll
  for (int e = 0; e < 16; ++e) ones[e] = (_Float16)1.0f;

  // Q as per-wave-constant B fragments (lane = query column i0+n).
  const _Float16* qrow = Qt + ((size_t)(b * 4096 + i0 + n)) * 64;
  const v16h qb0 = load_frag_b(qrow + koff);         // c 0..31
  const v16h qb1 = load_frag_b(qrow + 32 + koff);    // c 32..63

  v8f acc0 = {}, acc1 = {}, acc2 = {}, acc3 = {};    // Out[c, i]: rows c, col i = i0+n
  float m = -3.0e38f, l = 0.0f;                      // per-lane scalars (one query row)

  const _Float16* kbB = Kt + ((size_t)b * 4096) * 64;
  const _Float16* vB  = Vf + ((size_t)b * 64) * 4096;

  for (int j0 = 0; j0 < 4096; j0 += 128) {
    // ---- Phase A: S^T tiles (rows j, cols i) -- 16 WMMA ----
    v8f s[8];
#pragma unroll
    for (int t = 0; t < 8; ++t) {
      const _Float16* kr = kbB + (size_t)(j0 + t * 16) * 64;
      v16h ka0 = load_frag_a(kr, 64, lane);        // c 0..31
      v16h ka1 = load_frag_a(kr + 32, 64, lane);   // c 32..63
      v8f z = {};
      z    = WMMA_F16(ka0, qb0, z);
      s[t] = WMMA_F16(ka1, qb1, z);
    }

    // Prefetch next chunk of K/V toward the WGP (data resides in L2).
    if (j0 + 128 < 4096) {
      __builtin_prefetch(kbB + (size_t)(j0 + 128 + lane * 4) * 64, 0, 3);
      __builtin_prefetch(vB + (size_t)(lane * 2) * 4096 + j0 + 128, 0, 3);
      __builtin_prefetch(vB + (size_t)(lane * 2 + 1) * 4096 + j0 + 128, 0, 3);
    }

    // ---- Phase B: running max for this lane's query (exp2 domain) ----
    float cm = s[0][0];
#pragma unroll
    for (int t = 0; t < 8; ++t)
#pragma unroll
      for (int r = 0; r < 8; ++r)
        if (t | r) cm = fmaxf(cm, s[t][r]);
    cm = fmaxf(cm, xswap16_f(cm));                 // combine with partner half
    const float mn    = fmaxf(m, cm);
    const float alpha = __builtin_amdgcn_exp2f(m - mn);
    m = mn;
#pragma unroll
    for (int r = 0; r < 8; ++r) {
      acc0[r] *= alpha; acc1[r] *= alpha;
      acc2[r] *= alpha; acc3[r] *= alpha;
    }
    l *= alpha;

    // ---- Phase C+D: P^T B-fragments built in-lane, PV + row-sum (20 WMMA) ----
    v8f lsum = {};
#pragma unroll
    for (int u = 0; u < 4; ++u) {
      const int t0 = 2 * u, t1 = 2 * u + 1;
      float p0[8], p1[8];
#pragma unroll
      for (int r = 0; r < 8; ++r) {
        p0[r] = __builtin_amdgcn_exp2f(s[t0][r] - m);
        p1[r] = __builtin_amdgcn_exp2f(s[t1][r] - m);
      }
      // Build P^T B-fragment (K = 32 keys of this sub-chunk, column = lane's query).
      // hl=0 owns j 0..7 (t0) & 16..23 (t1); hl=1 owns j 8..15 (t0) & 24..31 (t1).
      V16HU pb;
#pragma unroll
      for (int q = 0; q < 4; ++q) {
        const unsigned o0 = pack2h(p0[2 * q], p0[2 * q + 1]);   // tile t0 pair
        const unsigned o1 = pack2h(p1[2 * q], p1[2 * q + 1]);   // tile t1 pair
        const unsigned sendv = hl ? o0 : o1;                    // sent to partner; recv unused
        const unsigned recv  = xswap16_u(sendv);
        pb.u[q]     = o0;   // elements 0..7  = K 0..7  / 8..15
        pb.u[4 + q] = o1;   // elements 8..15 = K 16..23 / 24..31
      }
      lsum = WMMA_F16(ones, pb.h, lsum);
      const _Float16* vb = vB + (size_t)(j0 + u * 32);
      v16h va0 = load_frag_a(vb + (size_t)(0 * 16) * 4096, 4096, lane);
      v16h va1 = load_frag_a(vb + (size_t)(1 * 16) * 4096, 4096, lane);
      v16h va2 = load_frag_a(vb + (size_t)(2 * 16) * 4096, 4096, lane);
      v16h va3 = load_frag_a(vb + (size_t)(3 * 16) * 4096, 4096, lane);
      acc0 = WMMA_F16(va0, pb.h, acc0);
      acc1 = WMMA_F16(va1, pb.h, acc1);
      acc2 = WMMA_F16(va2, pb.h, acc2);
      acc3 = WMMA_F16(va3, pb.h, acc3);
    }
    l += lsum[0];                                  // rows of lsum are replicated
  }

  // Epilogue: out[b, c, i0+n] = acc / l ; lanes of a half write 16 consecutive floats.
  const float inv = 1.0f / l;
#pragma unroll
  for (int r = 0; r < 8; ++r) {
    const int c = r + (hl << 3);
    out[((size_t)(b * 64 +  0 + c)) * 4096 + i0 + n] = acc0[r] * inv;
    out[((size_t)(b * 64 + 16 + c)) * 4096 + i0 + n] = acc1[r] * inv;
    out[((size_t)(b * 64 + 32 + c)) * 4096 + i0 + n] = acc2[r] * inv;
    out[((size_t)(b * 64 + 48 + c)) * 4096 + i0 + n] = acc3[r] * inv;
  }
}

extern "C" void kernel_launch(void* const* d_in, const int* in_sizes, int n_in,
                              void* d_out, int out_size, void* d_ws, size_t ws_size,
                              hipStream_t stream) {
  (void)in_sizes; (void)n_in; (void)out_size; (void)ws_size;
  const float* x  = (const float*)d_in[0];
  // d_in[1] = t (unused by the reference computation)
  const float* Wq = (const float*)d_in[2];
  const float* bq = (const float*)d_in[3];
  const float* Wk = (const float*)d_in[4];
  const float* bk = (const float*)d_in[5];
  const float* Wv = (const float*)d_in[6];
  const float* bv = (const float*)d_in[7];

  _Float16* Qt = (_Float16*)d_ws;                    // [4,4096,64] f16 = 2 MB
  _Float16* Kt = Qt + (size_t)4 * 4096 * 64;         // [4,4096,64] f16 = 2 MB
  _Float16* Vf = Kt + (size_t)4 * 4096 * 64;         // [4,64,4096] f16 = 2 MB

  qkv_proj<<<dim3(256), dim3(256), 0, stream>>>(x, Wq, bq, Wk, bk, Wv, bv, Qt, Kt, Vf);
  flash_attn<<<dim3(128), dim3(256), 0, stream>>>(Qt, Kt, Vf, (float*)d_out);
}
